// LlamaDINT_EnhancedAttention_24592982737470
// MI455X (gfx1250) — matrix operand order repaired and checked by reading, not compiled, not run on hardware
//
#include <hip/hip_runtime.h>
#include <stdint.h>

// ---------------- problem constants ----------------
constexpr int S   = 2048;
constexpr int H   = 2048;
constexpr int NH  = 16;
constexpr int NKV = 8;
constexpr int HD  = 128;
constexpr int QKD = NH * HD;   // 2048
constexpr int KKD = NKV * HD;  // 1024
constexpr float SCALE = 0.08838834764831845f; // HD^-0.5
constexpr float NEGV  = -1.0e9f;

typedef __attribute__((ext_vector_type(16))) _Float16 v16h;
typedef __attribute__((ext_vector_type(8)))  _Float16 v8h;
typedef __attribute__((ext_vector_type(8)))  float    v8f;
typedef __attribute__((ext_vector_type(4)))  unsigned int u32x4;
typedef __attribute__((ext_vector_type(8)))  int      i32x8;
typedef __attribute__((ext_vector_type(4)))  int      i32x4;

// ---------------- TDM availability ----------------
#if defined(__has_builtin)
#  if __has_builtin(__builtin_amdgcn_tensor_load_to_lds) && \
      __has_builtin(__builtin_amdgcn_s_wait_tensorcnt)
#    define USE_TDM 1
#  endif
#endif
#ifndef USE_TDM
#  define USE_TDM 0
#endif

#if USE_TDM
// Issue a 2D TDM load: tile of 32 rows x 32 f16 elems (rows strided by ldb elems)
// from global into contiguous LDS at byte offset lds_off. D# per ISA 8.3/8.4.
__device__ __forceinline__ void tdm_issue_b(const _Float16* Bt, int ldb, int col0,
                                            int k0, unsigned lds_off) {
  unsigned long long ga =
      (unsigned long long)(const void*)(Bt + (size_t)col0 * ldb + k0);
  unsigned galo = (unsigned)__builtin_amdgcn_readfirstlane((int)(unsigned)ga);
  unsigned gahi = (unsigned)__builtin_amdgcn_readfirstlane((int)(unsigned)(ga >> 32));
  unsigned ld   = (unsigned)__builtin_amdgcn_readfirstlane((int)lds_off);
  u32x4 g0 = { 1u,                                   // count=1 (user descriptor)
               ld,                                   // lds_addr (bytes)
               galo,                                 // global_addr[31:0]
               (gahi & 0x01ffffffu) | (2u << 30) };  // global_addr[56:32] | type=2
  i32x8 g1;
  g1[0] = (int)(1u << 16);   // workgroup_mask=0, data_size=1 (2 bytes)
  g1[1] = (int)(32u << 16);  // tensor_dim0[15:0]=32 in bits[31:16]
  g1[2] = (int)(32u << 16);  // tensor_dim0 hi=0 | tensor_dim1[15:0]=32
  g1[3] = (int)(32u << 16);  // tensor_dim1 hi=0 | tile_dim0=32
  g1[4] = 32;                // tile_dim1=32, tile_dim2=0
  g1[5] = ldb;               // tensor_dim0_stride[31:0] (elements)
  g1[6] = 0;                 // stride hi | dim1_stride lo
  g1[7] = 0;
  i32x4 z4 = {0, 0, 0, 0};
#if __clang_major__ >= 23
  i32x8 z8 = {0, 0, 0, 0, 0, 0, 0, 0};
  __builtin_amdgcn_tensor_load_to_lds(g0, g1, z4, z4, z8, 0);
#else
  __builtin_amdgcn_tensor_load_to_lds(g0, g1, z4, z4, 0);
#endif
}
#endif

// ---------------- WMMA fragment loaders (contiguous k = 16*hsel + s; A and B share it) ----------------
__device__ __forceinline__ v16h load_a_h16(const _Float16* __restrict__ A,
                                           int lda, int row0, int k0) {
  const int lane = threadIdx.x & 31;
  const int r    = lane & 15;
  const int hsel = lane >> 4;
  const _Float16* p = A + (size_t)(row0 + r) * lda + k0 + hsel * 16;
  v8h lo = *(const v8h*)p;
  v8h hi = *(const v8h*)(p + 8);
  v16h out;
#pragma unroll
  for (int i = 0; i < 8; ++i) { out[i] = lo[i]; out[8 + i] = hi[i]; }
  return out;
}

__device__ __forceinline__ v16h load_a_f32(const float* __restrict__ A,
                                           int lda, int row0, int k0) {
  const int lane = threadIdx.x & 31;
  const int r    = lane & 15;
  const int hsel = lane >> 4;
  const float* p = A + (size_t)(row0 + r) * lda + k0 + hsel * 16;
  v8f lo = *(const v8f*)p;
  v8f hi = *(const v8f*)(p + 8);
  v16h out;
#pragma unroll
  for (int i = 0; i < 8; ++i) {
    out[i]     = (_Float16)lo[i];
    out[8 + i] = (_Float16)hi[i];
  }
  return out;
}

__device__ __forceinline__ v16h load_b_h16(const _Float16* __restrict__ Bt,
                                           int ldb, int col0, int k0) {
  const int lane = threadIdx.x & 31;
  const int n    = lane & 15;
  const int hsel = lane >> 4;
  const _Float16* p = Bt + (size_t)(col0 + n) * ldb + k0 + hsel * 16;
  return *(const v16h*)p;
}

__device__ __forceinline__ v8f wmma_f16(v16h a, v16h b, v8f c) {
  return __builtin_amdgcn_wmma_f32_16x16x32_f16(false, a, false, b,
                                                (short)0, c, false, false);
}

#if USE_TDM
// One k-step of the TDM-staged GEMM: A from global (overlaps in-flight TDM),
// then wait TENSORcnt<=WAITN, compiler memory barrier, then ds-reads of the
// staged B tile (plain loads -> ds_load_b128, wave scope).
template <int WAITN>
__device__ __forceinline__ void gemm_tdm_tile(const _Float16* __restrict__ A, int lda,
                                              int row0, int k0,
                                              const _Float16 (&bt)[32 * 32],
                                              int n, int hsel, v8f acc[4][2]) {
  v16h a[4];
#pragma unroll
  for (int i = 0; i < 4; ++i) a[i] = load_a_h16(A, lda, row0 + 16 * i, k0);
  __builtin_amdgcn_s_wait_tensorcnt(WAITN);
  asm volatile("" ::: "memory");  // no value forwarding across the TDM wait
  v16h b[2];
#pragma unroll
  for (int j = 0; j < 2; ++j)
    b[j] = *(const v16h*)(&bt[(16 * j + n) * 32 + hsel * 16]);
#pragma unroll
  for (int i = 0; i < 4; ++i)
#pragma unroll
    for (int j = 0; j < 2; ++j) acc[i][j] = wmma_f16(a[i], b[j], acc[i][j]);
}
#endif

// ---------------- elementwise helpers ----------------
__global__ void cvt_kernel(const float* __restrict__ in, _Float16* __restrict__ out,
                           int n8) {  // n8 = n/8 (all sizes divisible by 8)
  int i = blockIdx.x * blockDim.x + threadIdx.x;
  if (i >= n8) return;
  v8f x = ((const v8f*)in)[i];
  v8h y;
#pragma unroll
  for (int j = 0; j < 8; ++j) y[j] = (_Float16)x[j];
  ((v8h*)out)[i] = y;
}

// LDS-tiled transpose+convert: in [R][C] (fp32 or f16) -> out f16 [C][R].
template <typename TIN>
__global__ __launch_bounds__(256) void transpose_cvt_kernel(
    const TIN* __restrict__ in, _Float16* __restrict__ out, int R, int C) {
  __shared__ _Float16 tile[32][33];
  const int tx = threadIdx.x & 31;
  const int ty = threadIdx.x >> 5;  // 0..7
  const int r0 = blockIdx.y * 32;
  const int c0 = blockIdx.x * 32;
#pragma unroll
  for (int i = 0; i < 4; ++i) {
    int r = ty + i * 8;
    tile[r][tx] = (_Float16)in[(size_t)(r0 + r) * C + c0 + tx];
  }
  __syncthreads();
#pragma unroll
  for (int i = 0; i < 4; ++i) {
    int c = ty + i * 8;
    out[(size_t)(c0 + c) * R + r0 + tx] = tile[tx][c];
  }
}

__global__ void zero_kernel(float* __restrict__ p, int n) {
  int i = blockIdx.x * blockDim.x + threadIdx.x;
  if (i < n) p[i] = 0.f;
}

// RoPE in place: X is [S][nheads*HD] f16; cos/sin are [S][HD] f32.
__global__ void rope_kernel(_Float16* __restrict__ X,
                            const float* __restrict__ cosb,
                            const float* __restrict__ sinb, int nheads) {
  int i = blockIdx.x * blockDim.x + threadIdx.x;
  int total = S * nheads * (HD / 2);
  if (i >= total) return;
  int d = i % (HD / 2);
  int t = i / (HD / 2);
  int h = t % nheads;
  int s = t / nheads;
  const int stride = nheads * HD;
  size_t i0 = (size_t)s * stride + h * HD + d;
  size_t i1 = i0 + HD / 2;
  float x0 = (float)X[i0], x1 = (float)X[i1];
  float c0 = cosb[s * HD + d],          s0 = sinb[s * HD + d];
  float c1 = cosb[s * HD + d + HD / 2], s1 = sinb[s * HD + d + HD / 2];
  X[i0] = (_Float16)(x0 * c0 - x1 * s0);
  X[i1] = (_Float16)(x1 * c1 + x0 * s1);
}

// ---------------- generic f16 WMMA GEMM ----------------
// C(MxN) = A(MxK f16, lda) * Bt(NxK f16, ldb)^T + bias; store f16 (Ch) or f32 (Cf).
// Block = 256 threads (8 waves), block tile 128x128, wave tile 64x32.
// B operand staged through LDS by the Tensor Data Mover (double-buffered,
// per-wave descriptors, TENSORcnt-synchronized) when available.
__global__ __launch_bounds__(256) void gemm_f16_kernel(
    const _Float16* __restrict__ A, const _Float16* __restrict__ Bt,
    _Float16* __restrict__ Ch, float* __restrict__ Cf,
    const float* __restrict__ bias,
    int M, int N, int K, int lda, int ldb, int ldc) {
  const int w    = threadIdx.x >> 5;
  const int wr   = w & 1;
  const int wc   = w >> 1;
  const int row0 = blockIdx.y * 128 + wr * 64;
  const int col0 = blockIdx.x * 128 + wc * 32;
  const int lane = threadIdx.x & 31;
  const int n    = lane & 15;
  const int hsel = lane >> 4;

  v8f acc[4][2] = {};

#if USE_TDM
  __shared__ _Float16 btile[8][2][32 * 32];  // 2 KB per wave per buffer
  // Statically-unprovable store of a runtime value: stops LLVM from treating
  // btile as store-free (it cannot see that tensor_load_to_lds writes it) and
  // undef-folding the reads.
  if (K < 0) btile[0][0][0] = (_Float16)(float)K;
  const unsigned base_off =
      (unsigned)((const char*)&btile[w][0][0] - (const char*)&btile[0][0][0]);
  tdm_issue_b(Bt, ldb, col0, 0, base_off);
  int cur = 0;
  int k0 = 0;
  for (; k0 + 32 < K; k0 += 32) {  // steady state: one TDM always in flight
    tdm_issue_b(Bt, ldb, col0, k0 + 32, base_off + (unsigned)((cur ^ 1) * 2048));
    gemm_tdm_tile<1>(A, lda, row0, k0, btile[w][cur], n, hsel, acc);
    cur ^= 1;
  }
  gemm_tdm_tile<0>(A, lda, row0, k0, btile[w][cur], n, hsel, acc);  // drain
#else
  for (int k0 = 0; k0 < K; k0 += 32) {
    if (k0 + 64 < K) {
      __builtin_prefetch(Bt + (size_t)(col0 + n) * ldb + k0 + 64, 0, 1);
      __builtin_prefetch(A + (size_t)(row0 + n) * lda + k0 + 64, 0, 1);
    }
    v16h a[4], b[2];
#pragma unroll
    for (int i = 0; i < 4; ++i) a[i] = load_a_h16(A, lda, row0 + 16 * i, k0);
#pragma unroll
    for (int j = 0; j < 2; ++j) b[j] = load_b_h16(Bt, ldb, col0 + 16 * j, k0);
#pragma unroll
    for (int i = 0; i < 4; ++i)
#pragma unroll
      for (int j = 0; j < 2; ++j) acc[i][j] = wmma_f16(a[i], b[j], acc[i][j]);
  }
#endif

  const int rb = hsel * 8;
#pragma unroll
  for (int i = 0; i < 4; ++i)
#pragma unroll
    for (int j = 0; j < 2; ++j) {
      int col = col0 + 16 * j + n;
      float bv = bias ? bias[col] : 0.f;
#pragma unroll
      for (int r = 0; r < 8; ++r) {
        int row = row0 + 16 * i + rb + r;
        float v = acc[i][j][r] + bv;
        if (Cf) Cf[(size_t)row * ldc + col] = v;
        else    Ch[(size_t)row * ldc + col] = (_Float16)v;
      }
    }
}

// ---------------- attention score / softmax / combine ----------------
__device__ __forceinline__ v8f score_tile(const v16h aq[4],
                                          const _Float16* __restrict__ Kb,
                                          int key0) {
  v8f acc = {};
#pragma unroll
  for (int kc = 0; kc < 4; ++kc)
    acc = wmma_f16(aq[kc], load_b_h16(Kb, KKD, key0, kc * 32), acc);
  return acc;
}

// PASS2=false: a1 = softmax(QK^T*scale+mask) -> Wout; colsum += column sums of a1.
// PASS2=true : a2 = softmax(Q2K2^T*scale+mask);
//              Wout = a1 + lam*colmean1 - lam*a2 (in place over a1).
template <bool PASS2>
__global__ __launch_bounds__(256) void attn_kernel(
    const _Float16* __restrict__ Q, const _Float16* __restrict__ Kt,
    float* __restrict__ Wout, float* __restrict__ colsum,
    const float* __restrict__ lambd) {
  const int h    = blockIdx.y;
  const int kv   = h >> 1;  // N_REP = 2
  const int wid  = threadIdx.x >> 5;
  const int q0   = blockIdx.x * 128 + wid * 16;
  const int lane = threadIdx.x & 31;
  const int n    = lane & 15;
  const int hsel = lane >> 4;
  const int rb   = hsel * 8;

  const _Float16* Qb = Q + h * HD;   // lda = QKD
  const _Float16* Kb = Kt + kv * HD; // ldb = KKD

  v16h aq[4];
#pragma unroll
  for (int kc = 0; kc < 4; ++kc) aq[kc] = load_a_h16(Qb, QKD, q0, kc * 32);

  const int ntc = q0 / 16 + 1;  // tiles with at least one unmasked key

  float m[8], ssum[8];
#pragma unroll
  for (int r = 0; r < 8; ++r) { m[r] = -3.0e38f; ssum[r] = 0.f; }

  // sweep 1: row max
  for (int t = 0; t < ntc; ++t) {
    v8f acc = score_tile(aq, Kb, t * 16);
    int key = t * 16 + n;
#pragma unroll
    for (int r = 0; r < 8; ++r) {
      int qq = q0 + rb + r;
      float x = (key <= qq) ? acc[r] * SCALE : NEGV;
      m[r] = fmaxf(m[r], x);
    }
  }
#pragma unroll
  for (int r = 0; r < 8; ++r)
    for (int off = 1; off < 16; off <<= 1)
      m[r] = fmaxf(m[r], __shfl_xor(m[r], off, 32));

  // sweep 2: exp sum
  for (int t = 0; t < ntc; ++t) {
    v8f acc = score_tile(aq, Kb, t * 16);
    int key = t * 16 + n;
#pragma unroll
    for (int r = 0; r < 8; ++r) {
      int qq = q0 + rb + r;
      float x = (key <= qq) ? acc[r] * SCALE : NEGV;
      ssum[r] += __expf(x - m[r]);
    }
  }
  float rinv[8];
#pragma unroll
  for (int r = 0; r < 8; ++r) {
    float s = ssum[r];
    for (int off = 1; off < 16; off <<= 1) s += __shfl_xor(s, off, 32);
    rinv[r] = 1.0f / s;
  }

  const float lam  = PASS2 ? lambd[0] : 0.f;
  const float invS = 1.0f / (float)S;
  float* Wh = Wout + (size_t)h * S * S;

  // sweep 3: write probs (pass1) or fused combine (pass2)
  for (int t = 0; t < S / 16; ++t) {
    int key = t * 16 + n;
    float p[8];
    if (t < ntc) {
      v8f acc = score_tile(aq, Kb, t * 16);
#pragma unroll
      for (int r = 0; r < 8; ++r) {
        int qq = q0 + rb + r;
        float x = (key <= qq) ? acc[r] * SCALE : NEGV;
        p[r] = __expf(x - m[r]) * rinv[r];
      }
    } else {
#pragma unroll
      for (int r = 0; r < 8; ++r) p[r] = 0.f;
    }
    if (!PASS2) {
      float cs = 0.f;
#pragma unroll
      for (int r = 0; r < 8; ++r) {
        Wh[(size_t)(q0 + rb + r) * S + key] = p[r];
        cs += p[r];
      }
      cs += __shfl_xor(cs, 16, 32);
      if (hsel == 0 && t < ntc) atomicAdd(&colsum[h * S + key], cs);
    } else {
      float cm = colsum[h * S + key] * invS * lam;
#pragma unroll
      for (int r = 0; r < 8; ++r) {
        size_t idx = (size_t)(q0 + rb + r) * S + key;
        float a1 = (t < ntc) ? Wh[idx] : 0.f;
        Wh[idx] = a1 + cm - lam * p[r];
      }
    }
  }
}

// ---------------- O_h = W_h @ V  (A is fp32 probs, converted on the fly) ----------------
__global__ __launch_bounds__(256) void av_gemm_kernel(
    const float* __restrict__ W, const _Float16* __restrict__ Vt,
    _Float16* __restrict__ Oh) {
  const int h    = blockIdx.y;
  const int kv   = h >> 1;
  const int w    = threadIdx.x >> 5;
  const int wr   = w & 1;
  const int wc   = w >> 1;
  const int row0 = blockIdx.x * 128 + wr * 64;
  const int col0 = wc * 32;  // N = HD = 128 covered by 4 column-waves
  const float* Ab = W + (size_t)h * S * S;       // lda = S
  const _Float16* Bt = Vt + (size_t)kv * HD * S; // [HD][S], ldb = S

  v8f acc[4][2] = {};
  for (int k0 = 0; k0 < S; k0 += 32) {
    v16h a[4], b[2];
#pragma unroll
    for (int i = 0; i < 4; ++i) a[i] = load_a_f32(Ab, S, row0 + 16 * i, k0);
#pragma unroll
    for (int j = 0; j < 2; ++j) b[j] = load_b_h16(Bt, S, col0 + 16 * j, k0);
#pragma unroll
    for (int i = 0; i < 4; ++i)
#pragma unroll
      for (int j = 0; j < 2; ++j) acc[i][j] = wmma_f16(a[i], b[j], acc[i][j]);
  }
  const int lane = threadIdx.x & 31;
  const int n = lane & 15, rb = (lane >> 4) * 8;
#pragma unroll
  for (int i = 0; i < 4; ++i)
#pragma unroll
    for (int j = 0; j < 2; ++j)
#pragma unroll
      for (int r = 0; r < 8; ++r) {
        int row = row0 + 16 * i + rb + r;
        int col = h * HD + col0 + 16 * j + n;
        Oh[(size_t)row * QKD + col] = (_Float16)acc[i][j][r];
      }
}

// ---------------- host orchestration ----------------
extern "C" void kernel_launch(void* const* d_in, const int* in_sizes, int n_in,
                              void* d_out, int out_size, void* d_ws, size_t ws_size,
                              hipStream_t stream) {
  (void)in_sizes; (void)n_in; (void)out_size; (void)ws_size;
  const float* hidden = (const float*)d_in[0];
  const float* cosb   = (const float*)d_in[1];
  const float* sinb   = (const float*)d_in[2];
  // d_in[3] = attention_mask (causal tril; computed analytically on device)
  const float* wq     = (const float*)d_in[4];
  const float* wk     = (const float*)d_in[5];
  const float* wv     = (const float*)d_in[6];
  const float* wo     = (const float*)d_in[7];
  const float* wq2a   = (const float*)d_in[8];
  const float* bq2a   = (const float*)d_in[9];
  const float* wq2b   = (const float*)d_in[10];
  const float* bq2b   = (const float*)d_in[11];
  const float* wk2a   = (const float*)d_in[12];
  const float* bk2a   = (const float*)d_in[13];
  const float* wk2b   = (const float*)d_in[14];
  const float* bk2b   = (const float*)d_in[15];
  const float* lambd  = (const float*)d_in[16];

  float* out_attn = (float*)d_out;                    // [S][H]
  float* out_w    = (float*)d_out + (size_t)S * H;    // [NH][S][S]

  // workspace carving (256B aligned)
  char* wp = (char*)d_ws;
  auto take = [&](size_t elems, size_t esz) -> void* {
    void* r = (void*)wp;
    wp += (elems * esz + 255) & ~(size_t)255;
    return r;
  };
  _Float16* Xh     = (_Float16*)take((size_t)S * H, 2);
  _Float16* wqT    = (_Float16*)take((size_t)H * QKD, 2);
  _Float16* wkT    = (_Float16*)take((size_t)H * KKD, 2);
  _Float16* wvT    = (_Float16*)take((size_t)H * KKD, 2);
  _Float16* woT    = (_Float16*)take((size_t)QKD * H, 2);
  _Float16* wq2aT  = (_Float16*)take((size_t)H * 1024, 2);
  _Float16* wq2bT  = (_Float16*)take((size_t)1024 * 2048, 2);
  _Float16* wk2aT  = (_Float16*)take((size_t)H * 512, 2);
  _Float16* wk2bT  = (_Float16*)take((size_t)512 * 1024, 2);
  _Float16* Qh     = (_Float16*)take((size_t)S * QKD, 2);
  _Float16* Kh     = (_Float16*)take((size_t)S * KKD, 2);
  _Float16* Vh     = (_Float16*)take((size_t)S * KKD, 2);
  _Float16* Q2a    = (_Float16*)take((size_t)S * 1024, 2);
  _Float16* Q2h    = (_Float16*)take((size_t)S * QKD, 2);
  _Float16* K2a    = (_Float16*)take((size_t)S * 512, 2);
  _Float16* K2h    = (_Float16*)take((size_t)S * KKD, 2);
  _Float16* Vt     = (_Float16*)take((size_t)KKD * S, 2);
  _Float16* Oh     = (_Float16*)take((size_t)S * QKD, 2);
  float*    colsum = (float*)take((size_t)NH * S, 4);

  auto blocks = [](size_t n) { return (unsigned)((n + 255) / 256); };
  auto tgrid  = [](int R, int C) { return dim3((unsigned)(C / 32), (unsigned)(R / 32)); };

  // 1) convert + LDS-tiled transposes
  cvt_kernel<<<blocks((size_t)S * H / 8), 256, 0, stream>>>(hidden, Xh, S * H / 8);
  transpose_cvt_kernel<float><<<tgrid(H, QKD), 256, 0, stream>>>(wq, wqT, H, QKD);
  transpose_cvt_kernel<float><<<tgrid(H, KKD), 256, 0, stream>>>(wk, wkT, H, KKD);
  transpose_cvt_kernel<float><<<tgrid(H, KKD), 256, 0, stream>>>(wv, wvT, H, KKD);
  transpose_cvt_kernel<float><<<tgrid(QKD, H), 256, 0, stream>>>(wo, woT, QKD, H);
  transpose_cvt_kernel<float><<<tgrid(H, 1024), 256, 0, stream>>>(wq2a, wq2aT, H, 1024);
  transpose_cvt_kernel<float><<<tgrid(1024, 2048), 256, 0, stream>>>(wq2b, wq2bT, 1024, 2048);
  transpose_cvt_kernel<float><<<tgrid(H, 512), 256, 0, stream>>>(wk2a, wk2aT, H, 512);
  transpose_cvt_kernel<float><<<tgrid(512, 1024), 256, 0, stream>>>(wk2b, wk2bT, 512, 1024);
  zero_kernel<<<blocks((size_t)NH * S), 256, 0, stream>>>(colsum, NH * S);

  auto gemm = [&](const _Float16* A, const _Float16* Bt, _Float16* Ch, float* Cf,
                  const float* bias, int M, int N, int K, int lda, int ldb, int ldc) {
    dim3 g(N / 128, M / 128);
    gemm_f16_kernel<<<g, 256, 0, stream>>>(A, Bt, Ch, Cf, bias, M, N, K, lda, ldb, ldc);
  };

  // 2) projections
  gemm(Xh, wqT,   Qh,  nullptr, nullptr, S, QKD, H, H, H, QKD);
  gemm(Xh, wkT,   Kh,  nullptr, nullptr, S, KKD, H, H, H, KKD);
  gemm(Xh, wvT,   Vh,  nullptr, nullptr, S, KKD, H, H, H, KKD);
  gemm(Xh, wq2aT, Q2a, nullptr, bq2a,    S, 1024, H, H, H, 1024);
  gemm(Q2a, wq2bT, Q2h, nullptr, bq2b,   S, 2048, 1024, 1024, 1024, 2048);
  gemm(Xh, wk2aT, K2a, nullptr, bk2a,    S, 512, H, H, H, 512);
  gemm(K2a, wk2bT, K2h, nullptr, bk2b,   S, 1024, 512, 512, 512, 1024);

  // 3) RoPE
  rope_kernel<<<blocks((size_t)S * NH * (HD / 2)), 256, 0, stream>>>(Qh, cosb, sinb, NH);
  rope_kernel<<<blocks((size_t)S * NKV * (HD / 2)), 256, 0, stream>>>(Kh, cosb, sinb, NKV);
  rope_kernel<<<blocks((size_t)S * NH * (HD / 2)), 256, 0, stream>>>(Q2h, cosb, sinb, NH);
  rope_kernel<<<blocks((size_t)S * NKV * (HD / 2)), 256, 0, stream>>>(K2h, cosb, sinb, NKV);

  // 4) V transpose for contiguous B fragments: [S][KKD] -> [KKD][S]
  transpose_cvt_kernel<_Float16><<<tgrid(S, KKD), 256, 0, stream>>>(Vh, Vt, S, KKD);

  // 5) attention maps (pass1 a1 + colsums, pass2 fused combine into d_out)
  {
    dim3 g(S / 128, NH);
    attn_kernel<false><<<g, 256, 0, stream>>>(Qh, Kh, out_w, colsum, nullptr);
    attn_kernel<true ><<<g, 256, 0, stream>>>(Q2h, K2h, out_w, colsum, lambd);
  }

  // 6) O = W @ V (per head), then output projection
  {
    dim3 g(S / 128, NH);
    av_gemm_kernel<<<g, 256, 0, stream>>>(out_w, Vt, Oh);
  }
  gemm(Oh, woT, nullptr, out_attn, nullptr, S, H, QKD, QKD, QKD, H);
}
